// STFT_38809324487458
// MI455X (gfx1250) — matrix operand order repaired and checked by reading, not compiled, not run on hardware
//
#include <hip/hip_runtime.h>
#include <math.h>

// ---------------------------------------------------------------------------
// STFT (32 x 262144, filter 1024, hop 256) as a split-precision f16 WMMA GEMM.
//   M = 513 real rows (+ paired imag rows at +513), K = 1024, N = 32*1025.
// Each wave32 computes a 16x32 tile of (real,imag) with fused magnitude/phase.
// Split precision: v = hi(f16) + lo(f16); v*w = hi*hi + hi*lo + lo*hi, fp32 acc
// -> ~2^-22 relative error (fp32 quality) on the matrix pipe.
// ---------------------------------------------------------------------------

typedef _Float16 half_t;
typedef __attribute__((ext_vector_type(8)))  _Float16 v8h;
typedef __attribute__((ext_vector_type(16))) _Float16 v16h;
typedef __attribute__((ext_vector_type(8)))  float    v8f;

#define FILTER_LEN 1024
#define HOP        256
#define CUTOFF     513
#define SIG_LEN    262144
#define PAD        512
#define PADDED_LEN (SIG_LEN + 2 * PAD)       // 263168
#define BATCH      32
#define FRAMES     1025                      // (PADDED_LEN - FILTER_LEN)/HOP + 1

#define BASIS_ELEMS (1026 * 1024)            // 1,050,624
#define X_ELEMS     (BATCH * PADDED_LEN)     // 8,421,376

// Workspace layout (bytes): basisHi | basisLo | xHi | xLo  (~38 MB total)
#define WS_BASIS_HI 0
#define WS_BASIS_LO ((size_t)BASIS_ELEMS * 2)
#define WS_X_HI     ((size_t)BASIS_ELEMS * 4)
#define WS_X_LO     (WS_X_HI + (size_t)X_ELEMS * 2)

// -------------------- pre-kernels: split fp32 -> f16 hi/lo -----------------

__global__ void stft_split_basis(const float* __restrict__ basis,
                                 half_t* __restrict__ hi,
                                 half_t* __restrict__ lo) {
  int i = blockIdx.x * 256 + threadIdx.x;
  if (i < BASIS_ELEMS) {
    float v = basis[i];
    half_t h = (half_t)v;
    hi[i] = h;
    lo[i] = (half_t)(v - (float)h);   // remainder is exactly representable
  }
}

__global__ void stft_split_signal(const float* __restrict__ x,
                                  half_t* __restrict__ hi,
                                  half_t* __restrict__ lo) {
  int i = blockIdx.x * 256 + threadIdx.x;   // over BATCH*PADDED_LEN
  if (i >= X_ELEMS) return;
  int b = i / PADDED_LEN;
  int p = i - b * PADDED_LEN;
  int pos = p - PAD;                         // reflect padding (np.pad 'reflect')
  pos = pos < 0 ? -pos : pos;
  if (pos >= SIG_LEN) pos = 2 * SIG_LEN - 2 - pos;
  float v = x[(size_t)b * SIG_LEN + pos];
  half_t h = (half_t)v;
  hi[i] = h;
  lo[i] = (half_t)(v - (float)h);
}

// -------------------- fragment loaders (gfx1250 WMMA layouts) --------------

// A (16x32 f16): lane L<16 -> row M=L, halves 0..15 = K k0..k0+15 (contiguous);
//                lane L>=16 -> same row, K base shifted by +16. Two b128 loads.
__device__ __forceinline__ v16h load_a_frag(const half_t* __restrict__ row, int kbase) {
  v8h lo = *(const v8h*)(row + kbase);
  v8h hi = *(const v8h*)(row + kbase + 8);
  return __builtin_shufflevector(lo, hi, 0, 1, 2, 3, 4, 5, 6, 7,
                                          8, 9, 10, 11, 12, 13, 14, 15);
}

// B (32x16 f16): lane L<16 -> col N=L, halves 0..15 = K k0..k0+15 (contiguous);
//                lane L>=16 -> col L-16, K base +16. Two b128 loads.
__device__ __forceinline__ v16h load_b_frag(const half_t* __restrict__ p) {
  v8h lo = *(const v8h*)(p);
  v8h hi = *(const v8h*)(p + 8);
  return __builtin_shufflevector(lo, hi, 0, 1, 2, 3, 4, 5, 6, 7,
                                          8, 9, 10, 11, 12, 13, 14, 15);
}

__device__ __forceinline__ v8f wmma16(v16h a, v16h b, v8f c) {
  return __builtin_amdgcn_wmma_f32_16x16x32_f16(false, a, false, b,
                                                (short)0, c, false, false);
}

// -------------------- main GEMM + magnitude/phase epilogue -----------------

__global__ __launch_bounds__(32)
void stft_wmma_kernel(const half_t* __restrict__ bHi, const half_t* __restrict__ bLo,
                      const half_t* __restrict__ xHi, const half_t* __restrict__ xLo,
                      float* __restrict__ out) {
  const int lane   = threadIdx.x;      // 0..31, full wave => EXEC all-ones
  const int hiHalf = lane >> 4;        // 0: lanes 0-15, 1: lanes 16-31
  const int l15    = lane & 15;

  const int m0 = blockIdx.y * 16;      // real-row tile base (0..512)
  const int n0 = blockIdx.x * 32;      // frame tile base
  const int b  = blockIdx.z;           // batch

  // --- A row pointers (clamped branchlessly so EXEC stays full) ---
  int mReal = m0 + l15; if (mReal > 512) mReal = 512;          // rows < 513
  const half_t* aReHiP = bHi + (size_t)mReal * FILTER_LEN;
  const half_t* aReLoP = bLo + (size_t)mReal * FILTER_LEN;
  const half_t* aImHiP = bHi + (size_t)(mReal + 513) * FILTER_LEN; // <= 1025
  const half_t* aImLoP = bLo + (size_t)(mReal + 513) * FILTER_LEN;
  const int aOff = hiHalf * 16;

  // --- B column pointers: frame t covers x_pad[t*256 .. t*256+1023] ---
  int t0 = n0 + l15;      if (t0 > FRAMES - 1) t0 = FRAMES - 1;
  int t1 = n0 + 16 + l15; if (t1 > FRAMES - 1) t1 = FRAMES - 1;
  const size_t xb = (size_t)b * PADDED_LEN;
  const int bOff = hiHalf * 16;
  const half_t* b0HiP = xHi + xb + (size_t)t0 * HOP + bOff;
  const half_t* b0LoP = xLo + xb + (size_t)t0 * HOP + bOff;
  const half_t* b1HiP = xHi + xb + (size_t)t1 * HOP + bOff;
  const half_t* b1LoP = xLo + xb + (size_t)t1 * HOP + bOff;

  v8f accRe0 = {}, accIm0 = {}, accRe1 = {}, accIm1 = {};

  for (int k0 = 0; k0 < FILTER_LEN; k0 += 32) {
    // prefetch B hi-stream two K-steps ahead (emits global_prefetch_b8)
    __builtin_prefetch(b0HiP + k0 + 64, 0, 1);
    __builtin_prefetch(b1HiP + k0 + 64, 0, 1);

    const int ak = k0 + aOff;
    v16h aReH = load_a_frag(aReHiP, ak);
    v16h aReL = load_a_frag(aReLoP, ak);
    v16h aImH = load_a_frag(aImHiP, ak);
    v16h aImL = load_a_frag(aImLoP, ak);
    v16h c0H  = load_b_frag(b0HiP + k0);
    v16h c0L  = load_b_frag(b0LoP + k0);
    v16h c1H  = load_b_frag(b1HiP + k0);
    v16h c1L  = load_b_frag(b1LoP + k0);

    // split-precision product: hi*lo + lo*hi + hi*hi  (fp32 accumulate)
    accRe0 = wmma16(aReH, c0L, accRe0);
    accRe0 = wmma16(aReL, c0H, accRe0);
    accRe0 = wmma16(aReH, c0H, accRe0);

    accIm0 = wmma16(aImH, c0L, accIm0);
    accIm0 = wmma16(aImL, c0H, accIm0);
    accIm0 = wmma16(aImH, c0H, accIm0);

    accRe1 = wmma16(aReH, c1L, accRe1);
    accRe1 = wmma16(aReL, c1H, accRe1);
    accRe1 = wmma16(aReH, c1H, accRe1);

    accIm1 = wmma16(aImH, c1L, accIm1);
    accIm1 = wmma16(aImL, c1H, accIm1);
    accIm1 = wmma16(aImH, c1H, accIm1);
  }

  // --- fused epilogue: magnitude + phase ---
  // C/D layout: lanes 0-15 -> N=lane, VGPR r -> M=r; lanes 16-31 -> M=8+r.
  const size_t OUT_PLANE = (size_t)CUTOFF * FRAMES;        // 525,825
  const size_t PHASE_OFF = (size_t)BATCH * OUT_PLANE;      // 16,826,400
  const int col0 = n0 + l15;
  const int col1 = n0 + 16 + l15;

  #pragma unroll
  for (int r = 0; r < 8; ++r) {
    const int row = m0 + r + hiHalf * 8;
    if (row < CUTOFF) {
      const size_t rowBase = (size_t)b * OUT_PLANE + (size_t)row * FRAMES;
      if (col0 < FRAMES) {
        float re = accRe0[r], im = accIm0[r];
        size_t o = rowBase + col0;
        out[o]             = sqrtf(re * re + im * im);
        out[PHASE_OFF + o] = atan2f(im, re);
      }
      if (col1 < FRAMES) {
        float re = accRe1[r], im = accIm1[r];
        size_t o = rowBase + col1;
        out[o]             = sqrtf(re * re + im * im);
        out[PHASE_OFF + o] = atan2f(im, re);
      }
    }
  }
}

// -------------------------------- launch -----------------------------------

extern "C" void kernel_launch(void* const* d_in, const int* in_sizes, int n_in,
                              void* d_out, int out_size, void* d_ws, size_t ws_size,
                              hipStream_t stream) {
  const float* x     = (const float*)d_in[0];   // (32, 262144)
  const float* basis = (const float*)d_in[1];   // (1026, 1, 1024)
  float* out = (float*)d_out;                   // mag | phase, each 32*513*1025

  char* ws = (char*)d_ws;
  half_t* basisHi = (half_t*)(ws + WS_BASIS_HI);
  half_t* basisLo = (half_t*)(ws + WS_BASIS_LO);
  half_t* xHi     = (half_t*)(ws + WS_X_HI);
  half_t* xLo     = (half_t*)(ws + WS_X_LO);

  stft_split_basis<<<(BASIS_ELEMS + 255) / 256, 256, 0, stream>>>(basis, basisHi, basisLo);
  stft_split_signal<<<(X_ELEMS + 255) / 256, 256, 0, stream>>>(x, xHi, xLo);

  dim3 grid((FRAMES + 31) / 32,      // 33 frame tiles of 32
            (CUTOFF + 15) / 16,      // 33 row tiles of 16
            BATCH);                  // 32
  stft_wmma_kernel<<<grid, 32, 0, stream>>>(basisHi, basisLo, xHi, xLo, out);
}
